// Model_29987461661222
// MI455X (gfx1250) — compile-verified, matrix operand order checked by reading, not run on hardware
//
#include <hip/hip_runtime.h>
#include <hip/hip_bf16.h>

typedef __attribute__((ext_vector_type(16))) _Float16 v16h;
typedef __attribute__((ext_vector_type(8)))  float    v8f;

#define BB 16
#define LL 2048
#define DD 256
#define NN 64

// Map K index (0..31) to the f16 element slot required by the
// v_wmma_f32_16x16x32_f16 operand layout (ISA 7.12.2):
//   lane-group 0: elements 0..7 -> K 0..7,  elements 8..15 -> K 16..23
//   lane-group 1: elements 0..7 -> K 8..15, elements 8..15 -> K 24..31
__device__ __forceinline__ int slot_of_k(int k) {
    return ((k >> 1) & 8) + (k & 7) + ((k & 8) << 1);
}

// CDNA5 async global->LDS copy (VGLOBAL GLOBAL_LOAD_ASYNC_TO_LDS_B128,
// tracked by ASYNCcnt). ldsAddr is the low-32 LDS byte address (flat LDS
// addresses carry the offset in [31:0] per the aperture mapping).
__device__ __forceinline__ void async_g2l_b128(unsigned ldsAddr, const void* gaddr) {
    asm volatile("global_load_async_to_lds_b128 %0, %1, off"
                 :: "v"(ldsAddr), "v"((unsigned long long)(uintptr_t)gaddr)
                 : "memory");
}
__device__ __forceinline__ void wait_async0() {
    asm volatile("s_wait_asynccnt 0x0" ::: "memory");
}

// ---------------------------------------------------------------- embedding
__global__ __launch_bounds__(256) void embed_kernel(
    const float* __restrict__ x, const float* __restrict__ Wemb,
    const float* __restrict__ bemb, float* __restrict__ h)
{
    size_t idx = (size_t)blockIdx.x * 256 + threadIdx.x;
    int d = (int)(idx & (DD - 1));
    size_t bl = idx >> 8;
    h[idx] = fmaf(x[bl], Wemb[d], bemb[d]);
}

// -------------------------------------------- per-layer SSM discretization
__global__ __launch_bounds__(256) void ssm_setup_kernel(
    const float* __restrict__ log_dt, const float* __restrict__ logAre,
    const float* __restrict__ Aim,    const float* __restrict__ Cre,
    const float* __restrict__ Cim,
    float* __restrict__ wre, float* __restrict__ wim,
    float* __restrict__ ca,  float* __restrict__ cb)
{
    int idx = blockIdx.x * 256 + threadIdx.x;   // d*NN + n
    int d = idx >> 6;
    float dt  = __expf(log_dt[d]);
    float are = -__expf(logAre[idx]);
    float aim = Aim[idx];
    float er  = __expf(dt * are);
    float sc, cc;
    __sincosf(dt * aim, &sc, &cc);
    float wr = er * cc, wi = er * sc;
    float nr = wr - 1.0f, ni = wi;
    float den = fmaf(are, are, aim * aim);
    float inv = 1.0f / den;
    float fr = (nr * are + ni * aim) * inv;
    float fi = (ni * are - nr * aim) * inv;
    float cr = Cre[idx], ci = Cim[idx];
    float ctr = cr * fr - ci * fi;
    float cti = cr * fi + ci * fr;
    wre[idx] = wr; wim[idx] = wi;
    ca[idx] = 2.0f * ctr; cb[idx] = -2.0f * cti;
}

// ------------------------------------- BN(eval) + transpose [B,L,D]->[B,D,L]
__global__ __launch_bounds__(256) void bn_transpose_kernel(
    const float* __restrict__ h, const float* __restrict__ gamma,
    const float* __restrict__ beta, float* __restrict__ u)
{
    __shared__ float tile[32][33];
    int b = blockIdx.z, d0 = blockIdx.x * 32, l0 = blockIdx.y * 32;
    int tx = threadIdx.x, ty = threadIdx.y;
    #pragma unroll
    for (int q = 0; q < 4; ++q) {
        int l = ty + q * 8;
        tile[l][tx] = h[((size_t)b * LL + l0 + l) * DD + d0 + tx];
    }
    __syncthreads();
    #pragma unroll
    for (int q = 0; q < 4; ++q) {
        int di = ty + q * 8;
        float g = gamma[d0 + di], be = beta[d0 + di];
        u[((size_t)b * DD + d0 + di) * LL + l0 + tx] = fmaf(g, tile[tx][di], be);
    }
}

// ---------------------------------------------------------- diagonal SSM scan
__global__ __launch_bounds__(256) void ssm_scan_kernel(
    const float* __restrict__ u,
    const float* __restrict__ wre, const float* __restrict__ wim,
    const float* __restrict__ ca,  const float* __restrict__ cb,
    const float* __restrict__ Dskip, float* __restrict__ y)
{
    int lane = threadIdx.x & 31;
    int wave = threadIdx.x >> 5;
    int pair = blockIdx.x * 8 + wave;     // pair = b*DD + d
    int d = pair & (DD - 1);
    int n0 = d * NN + lane, n1 = n0 + 32;
    float w0r = wre[n0], w0i = wim[n0], a0 = ca[n0], b0 = cb[n0];
    float w1r = wre[n1], w1i = wim[n1], a1 = ca[n1], b1 = cb[n1];
    float dsk = Dskip[d];
    const float* up = u + (size_t)pair * LL;
    float*       yp = y + (size_t)pair * LL;
    float s0r = 0.f, s0i = 0.f, s1r = 0.f, s1i = 0.f;
    for (int l0 = 0; l0 < LL; l0 += 32) {
        float uv = up[l0 + lane];
        if (l0 + 256 < LL) __builtin_prefetch(up + l0 + 256, 0, 1);
        float yreg = 0.f;
        #pragma unroll
        for (int j = 0; j < 32; ++j) {
            float uu = __shfl(uv, j, 32);
            float t0r = fmaf(w0r, s0r, fmaf(-w0i, s0i, uu));
            float t0i = fmaf(w0r, s0i, w0i * s0r);
            float t1r = fmaf(w1r, s1r, fmaf(-w1i, s1i, uu));
            float t1i = fmaf(w1r, s1i, w1i * s1r);
            s0r = t0r; s0i = t0i; s1r = t1r; s1i = t1i;
            float part = fmaf(a0, s0r, fmaf(b0, s0i, fmaf(a1, s1r, b1 * s1i)));
            part += __shfl_xor(part, 16, 32);
            part += __shfl_xor(part,  8, 32);
            part += __shfl_xor(part,  4, 32);
            part += __shfl_xor(part,  2, 32);
            part += __shfl_xor(part,  1, 32);
            if (lane == j) yreg = fmaf(dsk, uu, part);
        }
        yp[l0 + lane] = yreg;
    }
}

// ---------------------------------- GELU(tanh) + transpose [B,D,L]->[B,L,D]
__device__ __forceinline__ float gelu_tanh(float v) {
    const float c = 0.7978845608028654f;
    float t = tanhf(c * fmaf(0.044715f * v * v, v, v));
    return 0.5f * v * (1.0f + t);
}

__global__ __launch_bounds__(256) void gelu_transpose_kernel(
    const float* __restrict__ y, float* __restrict__ g)
{
    __shared__ float tile[32][33];
    int b = blockIdx.z, d0 = blockIdx.x * 32, l0 = blockIdx.y * 32;
    int tx = threadIdx.x, ty = threadIdx.y;
    #pragma unroll
    for (int q = 0; q < 4; ++q) {
        int di = ty + q * 8;
        tile[di][tx] = y[((size_t)b * DD + d0 + di) * LL + l0 + tx];
    }
    __syncthreads();
    #pragma unroll
    for (int q = 0; q < 4; ++q) {
        int li = ty + q * 8;
        g[((size_t)b * LL + l0 + li) * DD + d0 + tx] = gelu_tanh(tile[tx][li]);
    }
}

// --------------------- f32 -> f16 weights, pre-permuted to WMMA tile layout
// For k-block kb (8 of 32) and n-block nb (4 of 64) the 32x64 tile is stored
// as a contiguous 4KB block: [n_local][slot_of_k(k_local)] f16 — exactly the
// LDS image the matmul needs, so B staging is one async b128 copy per thread.
__global__ __launch_bounds__(256) void convert_perm_kernel(
    const float* __restrict__ w, _Float16* __restrict__ wp)
{
    int i = blockIdx.x * 256 + threadIdx.x;   // k*256 + n
    int k = i >> 8, n = i & 255;
    int kb = k >> 5, kl = k & 31, nb = n >> 6, nl = n & 63;
    wp[(size_t)(kb * 4 + nb) * 2048 + nl * 32 + slot_of_k(kl)] = (_Float16)w[i];
}

// ------------------------------------------------ WMMA linear (+bias+res+act)
// out[M,256] = act(A[M,256] @ W[256,256] + bias [+ res])
// MODE 0: out = acc + bias + res
// MODE 1: out = relu(acc + bias)
// MODE 2: out = relu(acc + bias + res)
template<int MODE>
__global__ __launch_bounds__(256) void linear_wmma_kernel(
    const float* __restrict__ A, const _Float16* __restrict__ Wp,
    const float* __restrict__ bias, const float* __restrict__ res,
    float* __restrict__ out)
{
    __shared__ alignas(32) _Float16 As[2][128][32];   // fragment-slot layout
    __shared__ alignas(16) _Float16 Bs[2][64][32];    // async-copied, frag-ready
    __shared__ alignas(16) float    Cs[8][16][68];    // epilogue staging

    const int tid  = threadIdx.x;
    const int lane = tid & 31;
    const int wave = tid >> 5;
    const int rowBase = blockIdx.x * 128;
    const int colBase = blockIdx.y * 64;

    const int arow = tid >> 1;
    const int acb  = (tid & 1) * 16;
    const float* aRow = A + (size_t)(rowBase + arow) * 256 + acb;
    const _Float16* wTile = Wp + (size_t)blockIdx.y * 2048;   // + kb*4*2048

    auto stageA = [&](int kb, int buf) {
        const float4* src = (const float4*)(aRow + kb * 32);
        float4 f0 = src[0], f1 = src[1], f2 = src[2], f3 = src[3];
        float av[16] = { f0.x, f0.y, f0.z, f0.w, f1.x, f1.y, f1.z, f1.w,
                         f2.x, f2.y, f2.z, f2.w, f3.x, f3.y, f3.z, f3.w };
        #pragma unroll
        for (int e = 0; e < 16; ++e)
            As[buf][arow][slot_of_k(acb + e)] = (_Float16)av[e];
    };
    auto stageB = [&](int kb, int buf) {
        unsigned lds = (unsigned)(uintptr_t)(&Bs[buf][0][0]) + tid * 16;
        async_g2l_b128(lds, (const char*)(wTile + (size_t)kb * 8192) + tid * 16);
    };

    v8f acc[4] = {};
    stageA(0, 0);
    stageB(0, 0);

    for (int kb = 0; kb < 8; ++kb) {
        const int p = kb & 1;
        wait_async0();           // my async B(kb) landed (in-order, only one live)
        __syncthreads();         // everyone's A stores + async B for buf p done
        if (kb < 7) {            // prefetch next tile into the other buffer
            stageA(kb + 1, 1 - p);
            stageB(kb + 1, 1 - p);
        }
        const v16h af  = *(const v16h*)&As[p][wave * 16 + (lane & 15)][(lane >> 4) * 16];
        const v16h bf0 = *(const v16h*)&Bs[p][ 0 + (lane & 15)][(lane >> 4) * 16];
        const v16h bf1 = *(const v16h*)&Bs[p][16 + (lane & 15)][(lane >> 4) * 16];
        const v16h bf2 = *(const v16h*)&Bs[p][32 + (lane & 15)][(lane >> 4) * 16];
        const v16h bf3 = *(const v16h*)&Bs[p][48 + (lane & 15)][(lane >> 4) * 16];
        acc[0] = __builtin_amdgcn_wmma_f32_16x16x32_f16(false, af, false, bf0, (short)0, acc[0], false, false);
        acc[1] = __builtin_amdgcn_wmma_f32_16x16x32_f16(false, af, false, bf1, (short)0, acc[1], false, false);
        acc[2] = __builtin_amdgcn_wmma_f32_16x16x32_f16(false, af, false, bf2, (short)0, acc[2], false, false);
        acc[3] = __builtin_amdgcn_wmma_f32_16x16x32_f16(false, af, false, bf3, (short)0, acc[3], false, false);
    }

    // Spill accumulators to a wave-private LDS tile (no barrier needed: LDS
    // ops from one wave are in-order), then do vectorized bias/res/act I/O.
    #pragma unroll
    for (int t = 0; t < 4; ++t)
        #pragma unroll
        for (int r = 0; r < 8; ++r)
            Cs[wave][(lane >> 4) * 8 + r][t * 16 + (lane & 15)] = acc[t][r];

    const int orow  = lane >> 1;
    const int ocol  = (lane & 1) * 32;
    const int grow  = rowBase + wave * 16 + orow;
    const int gcol0 = colBase + ocol;
    const size_t gbase = (size_t)grow * 256 + gcol0;
    #pragma unroll
    for (int q = 0; q < 8; ++q) {
        float4 c  = *(const float4*)&Cs[wave][orow][ocol + q * 4];
        float4 bv = *(const float4*)&bias[gcol0 + q * 4];
        float4 rv = make_float4(0.f, 0.f, 0.f, 0.f);
        if (MODE != 1) rv = *(const float4*)&res[gbase + q * 4];
        float4 o;
        o.x = c.x + bv.x; o.y = c.y + bv.y; o.z = c.z + bv.z; o.w = c.w + bv.w;
        if (MODE == 0) { o.x += rv.x; o.y += rv.y; o.z += rv.z; o.w += rv.w; }
        if (MODE == 1) { o.x = fmaxf(o.x, 0.f); o.y = fmaxf(o.y, 0.f);
                         o.z = fmaxf(o.z, 0.f); o.w = fmaxf(o.w, 0.f); }
        if (MODE == 2) { o.x = fmaxf(o.x + rv.x, 0.f); o.y = fmaxf(o.y + rv.y, 0.f);
                         o.z = fmaxf(o.z + rv.z, 0.f); o.w = fmaxf(o.w + rv.w, 0.f); }
        *(float4*)&out[gbase + q * 4] = o;
    }
}

// --------------------------------------------------------------- mean + FC
__global__ __launch_bounds__(256) void mean_kernel(
    const float* __restrict__ h, float* __restrict__ meanb)
{
    int b = blockIdx.x, d = threadIdx.x;
    float s = 0.f;
    for (int l = 0; l < LL; ++l) s += h[((size_t)b * LL + l) * DD + d];
    meanb[b * DD + d] = s * (1.0f / LL);
}

__global__ void fc_kernel(const float* __restrict__ meanb,
                          const float* __restrict__ fcW,
                          const float* __restrict__ fcb,
                          float* __restrict__ out)
{
    int t = threadIdx.x;
    if (t >= 2 * BB) return;
    int b = t >> 1, o = t & 1;
    float s = fcb[o];
    for (int d = 0; d < DD; ++d) s = fmaf(meanb[b * DD + d], fcW[d * 2 + o], s);
    out[b * 2 + o] = s;
}

// ------------------------------------------------------------------- driver
extern "C" void kernel_launch(void* const* d_in, const int* in_sizes, int n_in,
                              void* d_out, int out_size, void* d_ws, size_t ws_size,
                              hipStream_t stream)
{
    (void)in_sizes; (void)n_in; (void)out_size; (void)ws_size;
    const float* x      = (const float*)d_in[0];
    const float* W_emb  = (const float*)d_in[1];
    const float* b_emb  = (const float*)d_in[2];
    const float* log_dt = (const float*)d_in[3];
    const float* logAre = (const float*)d_in[4];
    const float* Aim    = (const float*)d_in[5];
    const float* Cre    = (const float*)d_in[6];
    const float* Cim    = (const float*)d_in[7];
    const float* Dskip  = (const float*)d_in[8];
    const float* Wout   = (const float*)d_in[9];
    const float* bout   = (const float*)d_in[10];
    const float* gamma  = (const float*)d_in[11];
    const float* beta   = (const float*)d_in[12];
    const float* fW1    = (const float*)d_in[13];
    const float* fb1    = (const float*)d_in[14];
    const float* fW2    = (const float*)d_in[15];
    const float* fb2    = (const float*)d_in[16];
    const float* fcW    = (const float*)d_in[17];
    const float* fcb    = (const float*)d_in[18];
    float* outp = (float*)d_out;

    float* ws = (float*)d_ws;
    const size_t NBLD = (size_t)BB * LL * DD;      // 8,388,608 floats
    float* h   = ws;
    float* u   = ws + NBLD;
    float* y   = ws + 2 * NBLD;
    float* r   = ws + 3 * NBLD;
    float* wre = ws + 4 * NBLD;
    float* wim = wre + DD * NN;
    float* ca  = wim + DD * NN;
    float* cb  = ca  + DD * NN;
    float* meanb = cb + DD * NN;
    _Float16* w16 = (_Float16*)(meanb + BB * DD);

    const dim3 tGrid(DD / 32, LL / 32, BB);
    const dim3 tBlk(32, 8);
    const dim3 mmGrid((BB * LL) / 128, DD / 64);

    embed_kernel<<<(BB * LL * DD) / 256, 256, 0, stream>>>(x, W_emb, b_emb, h);

    auto s4d_layer = [&](int i) {
        ssm_setup_kernel<<<(DD * NN) / 256, 256, 0, stream>>>(
            log_dt + (size_t)i * DD, logAre + (size_t)i * DD * NN,
            Aim + (size_t)i * DD * NN, Cre + (size_t)i * DD * NN,
            Cim + (size_t)i * DD * NN, wre, wim, ca, cb);
        bn_transpose_kernel<<<tGrid, tBlk, 0, stream>>>(
            h, gamma + (size_t)i * DD, beta + (size_t)i * DD, u);
        ssm_scan_kernel<<<(BB * DD) / 8, 256, 0, stream>>>(
            u, wre, wim, ca, cb, Dskip + (size_t)i * DD, y);
        gelu_transpose_kernel<<<tGrid, tBlk, 0, stream>>>(y, u);
        convert_perm_kernel<<<(DD * DD) / 256, 256, 0, stream>>>(
            Wout + (size_t)i * DD * DD, w16);
        linear_wmma_kernel<0><<<mmGrid, 256, 0, stream>>>(
            u, w16, bout + (size_t)i * DD, h, h);
    };

    for (int i = 0; i < 4; ++i) s4d_layer(i);
    hipMemcpyAsync(r, h, NBLD * sizeof(float), hipMemcpyDeviceToDevice, stream);
    s4d_layer(4);
    s4d_layer(5);

    convert_perm_kernel<<<(DD * DD) / 256, 256, 0, stream>>>(fW1, w16);
    linear_wmma_kernel<1><<<mmGrid, 256, 0, stream>>>(h, w16, fb1, nullptr, y);
    convert_perm_kernel<<<(DD * DD) / 256, 256, 0, stream>>>(fW2, w16);
    linear_wmma_kernel<2><<<mmGrid, 256, 0, stream>>>(y, w16, fb2, r, h);

    mean_kernel<<<BB, 256, 0, stream>>>(h, meanb);
    fc_kernel<<<1, 64, 0, stream>>>(meanb, fcW, fcb, outp);
}
